// Performer_35888746726059
// MI455X (gfx1250) — matrix operand order repaired and checked by reading, not compiled, not run on hardware
//
#include <hip/hip_runtime.h>

// Performer (FAVOR+) attention for MI455X (gfx1250, wave32, WMMA).
// B=4,H=16 -> 64 heads; S=4096, D=64, M=64.
// All GEMMs via v_wmma_f32_16x16x32_bf16 (bf16 in, f32 accum).
// phi computed in exp2-domain: omega pre-scaled by log2(e), row term
// rh = -0.5*log2(e)*||x||^2 - 3 folds in the 1/sqrt(M)=2^-3 factor, so
// phi = exp2(p + rh) is a single v_add + native v_exp_f32 per element.

namespace {

constexpr int SEQ   = 4096;
constexpr int DIM   = 64;   // head dim D
constexpr int FM    = 64;   // feature count M
constexpr int ECOLS = 65;   // D + 1 (ones column) in buf1
constexpr int NHEAD = 64;   // B*H

constexpr float LOG2E  = 1.4426950408889634f;
constexpr float NH_L2E = -0.7213475204444817f;  // -0.5*log2(e)

typedef __attribute__((ext_vector_type(16))) __bf16 v16bf;
typedef __attribute__((ext_vector_type(8)))  float  v8f;

union Frag {
  unsigned short u[16];
  unsigned int   w[8];
  uint2          d2[4];
  uint4          q[2];
  v16bf          v;
};

// Native exp2 (v_exp_f32).
__device__ inline float exp2f_dev(float x) { return __builtin_amdgcn_exp2f(x); }

// Scalar f32->bf16 (RNE) -- only used in small staging loops.
__device__ inline unsigned short f2bf(float f) {
  unsigned x = __builtin_bit_cast(unsigned, f);
  x += 0x7FFFu + ((x >> 16) & 1u);
  return (unsigned short)(x >> 16);
}

// Hot-path pack of two f32 into packed bf16x2: one VALU instruction.
__device__ inline unsigned pack2(float a, float b) {
#if __has_builtin(__builtin_amdgcn_cvt_pk_bf16_f32)
  typedef __attribute__((ext_vector_type(2))) __bf16 v2bf;
  v2bf t = __builtin_amdgcn_cvt_pk_bf16_f32(a, b);   // v_cvt_pk_bf16_f32
  return __builtin_bit_cast(unsigned, t);
#else
  // v_perm_b32 truncation fallback: dst = {b[31:16], a[31:16]}
  return __builtin_amdgcn_perm(__builtin_bit_cast(unsigned, a),
                               __builtin_bit_cast(unsigned, b),
                               0x03020706u);
#endif
}

__device__ inline float sq4(float4 a) { return a.x*a.x + a.y*a.y + a.z*a.z + a.w*a.w; }
__device__ inline void pack8(Frag& f, int base, float4 a, float4 b) {
  f.w[base+0] = pack2(a.x, a.y); f.w[base+1] = pack2(a.z, a.w);
  f.w[base+2] = pack2(b.x, b.y); f.w[base+3] = pack2(b.z, b.w);
}

__device__ inline v8f wmma_bf16(const Frag& a, const Frag& b, v8f c) {
  return __builtin_amdgcn_wmma_f32_16x16x32_bf16(false, a.v, false, b.v,
                                                 (short)0, c, false, false);
}

// B fragment (32x16 bf16): lane holds column ncol, rows k+s (s<8) and
// k+16+(s-8) (s>=8), k = koff + 8*(lane>>4), from row-major [k][n] LDS storage.
__device__ inline void loadB(Frag& f, const unsigned short* base, int stride,
                             int k, int ncol) {
  const unsigned short* p = base + k * stride + ncol;
  #pragma unroll
  for (int s = 0; s < 16; ++s)
    f.u[s] = p[(s < 8 ? s : s + 8) * stride];
}

// A fragments for a 16x64 bf16 tile (two K=32 frags), loaded straight from a
// global f32 row (this lane's row = base row + lane&15; halves split columns).
// Also returns this lane's partial sum-of-squares (32 of the 64 elements).
__device__ inline float loadA_row(Frag a[2], const float* row, int hf) {
  float ss = 0.f;
  #pragma unroll
  for (int dc = 0; dc < 2; ++dc) {
    const float* p = row + 32*dc + 8*hf;
    float4 c0 = *reinterpret_cast<const float4*>(p);
    float4 c1 = *reinterpret_cast<const float4*>(p + 4);
    float4 c2 = *reinterpret_cast<const float4*>(p + 16);
    float4 c3 = *reinterpret_cast<const float4*>(p + 20);
    ss += sq4(c0) + sq4(c1) + sq4(c2) + sq4(c3);
    pack8(a[dc], 0, c0, c1);
    pack8(a[dc], 4, c2, c3);
  }
  return ss;
}

} // namespace

// ---------------------------------------------------------------------------
// Phase A: buf1[head] = phi(K)^T @ V  (64x64)  and  buf1[:,64] = rowsum(phi(K))
// grid (64 heads, 4 chunks) x 256 threads; each wave owns 128 rows (4 blocks
// of 32), accumulates full 64x65 in registers, LDS ds_add_f32 reduce, then
// global atomic add.
// ---------------------------------------------------------------------------
__global__ __launch_bounds__(256) void perfA(const float* __restrict__ Kg,
                                             const float* __restrict__ Vg,
                                             const float* __restrict__ Og,
                                             float* __restrict__ buf1) {
  constexpr int PH = 36;  // phi^T ushort row stride ([m=64][s=32+pad])
  __shared__ __align__(16) unsigned char pool[8192 + 8 * (64 * PH * 2)];
  unsigned short* omegaLds = reinterpret_cast<unsigned short*>(pool);
  float* redLds = reinterpret_cast<float*>(pool + 8192);  // aliases phi bufs

  const int tid  = threadIdx.x;
  const int lane = tid & 31;
  const int wv   = tid >> 5;
  const int r    = lane & 15;
  const int hf   = lane >> 4;
  const int head = blockIdx.x;

  unsigned short* phiT =
      reinterpret_cast<unsigned short*>(pool + 8192) + wv * (64 * PH);

  // stage omega*log2(e) bf16, row-major [d][m], stride 64 (exp2 domain)
  for (int idx = tid; idx < DIM * FM; idx += 256)
    omegaLds[idx] = f2bf(Og[idx] * LOG2E);
  __syncthreads();

  const v8f vzero = {0.f,0.f,0.f,0.f,0.f,0.f,0.f,0.f};
  v8f acc[4][4];
  #pragma unroll
  for (int a = 0; a < 4; ++a)
    #pragma unroll
    for (int b = 0; b < 4; ++b) acc[a][b] = vzero;
  float ksum[4] = {0.f, 0.f, 0.f, 0.f};

  const float* Kh = Kg + (size_t)head * SEQ * DIM;
  const float* Vh = Vg + (size_t)head * SEQ * DIM;
  const int sBase = blockIdx.y * 1024 + wv * 128;

  for (int blk = 0; blk < 4; ++blk) {
    const int s0 = sBase + blk * 32;

    // prefetch next block's K/V rows for this lane while we compute
    if (blk < 3) {
      __builtin_prefetch(Kh + (size_t)(s0 + 32 + r) * DIM + 8*hf, 0, 3);
      __builtin_prefetch(Vh + (size_t)(s0 + 32 + r + 16*hf) * DIM, 0, 3);
    }

    // ---- features: phi(K) for 32 rows, stored transposed [m][s] in LDS ----
    for (int sub = 0; sub < 2; ++sub) {
      Frag aK[2];
      float ss = loadA_row(aK, Kh + (size_t)(s0 + sub*16 + r) * DIM, hf);
      // -0.5*log2(e)*||k||^2 - 3   (the -3 folds in 1/sqrt(M) = 2^-3)
      float rr = fmaf(ss + __shfl_xor(ss, 16, 32), NH_L2E, -3.0f);
      float rh[8];
      #pragma unroll
      for (int i = 0; i < 8; ++i)      // broadcast, hoisted out of mt loop
        rh[i] = __shfl(rr, i + 8*hf, 32);
      #pragma unroll
      for (int mt = 0; mt < 4; ++mt) {
        Frag b0, b1;
        loadB(b0, omegaLds, FM,      8*hf, 16*mt + r);
        loadB(b1, omegaLds, FM, 32 + 8*hf, 16*mt + r);
        v8f p = vzero;
        p = wmma_bf16(aK[0], b0, p);
        p = wmma_bf16(aK[1], b1, p);
        float ph[8];
        #pragma unroll
        for (int i = 0; i < 8; ++i) {
          ph[i] = exp2f_dev(p[i] + rh[i]);  // one add + native v_exp_f32
          ksum[mt] += ph[i];                // ones-column contribution
        }
        unsigned short* dst = phiT + (16*mt + r) * PH + sub*16 + hf*8;
        *reinterpret_cast<uint2*>(dst) =
            make_uint2(pack2(ph[0], ph[1]), pack2(ph[2], ph[3]));
        *reinterpret_cast<uint2*>(dst + 4) =
            make_uint2(pack2(ph[4], ph[5]), pack2(ph[6], ph[7]));
      }
    }
    __syncthreads();

    // ---- acc += phi(K)^T (64 x s32) @ V (s32 x 64) ----
    Frag aT[4];
    #pragma unroll
    for (int mt = 0; mt < 4; ++mt) {  // A frag from row-major [m][s], stride 36
      const unsigned short* src = phiT + (16*mt + r) * PH + hf*8;
      aT[mt].d2[0] = *reinterpret_cast<const uint2*>(src);
      aT[mt].d2[1] = *reinterpret_cast<const uint2*>(src + 4);
      aT[mt].d2[2] = *reinterpret_cast<const uint2*>(src + 16);
      aT[mt].d2[3] = *reinterpret_cast<const uint2*>(src + 20);
    }
    const float* vrow = Vh + (size_t)(s0 + 8*hf) * DIM + r;  // this lane's V column
    #pragma unroll
    for (int et = 0; et < 4; ++et) {
      const float* p = vrow + 16*et;
      Frag bV;
      #pragma unroll
      for (int j = 0; j < 8; ++j) {
        int k0 = (j < 4) ? 2*j : 2*j + 8;
        bV.w[j] = pack2(p[k0 * DIM], p[(k0 + 1) * DIM]);
      }
      #pragma unroll
      for (int mt = 0; mt < 4; ++mt)
        acc[mt][et] = wmma_bf16(aT[mt], bV, acc[mt][et]);
    }
    __syncthreads();
  }

  // ---- cross-wave reduce in LDS (aliases dead phi buffers), then global ----
  for (int idx = tid; idx < FM * ECOLS; idx += 256) redLds[idx] = 0.f;
  __syncthreads();
  #pragma unroll
  for (int mt = 0; mt < 4; ++mt) {
    #pragma unroll
    for (int et = 0; et < 4; ++et)
      #pragma unroll
      for (int i = 0; i < 8; ++i)
        atomicAdd(&redLds[(16*mt + i + 8*hf) * ECOLS + 16*et + r],
                  acc[mt][et][i]);
    float ks = ksum[mt] + __shfl_xor(ksum[mt], 16, 32);
    if (hf == 0) atomicAdd(&redLds[(16*mt + r) * ECOLS + 64], ks);
  }
  __syncthreads();
  float* dst = buf1 + (size_t)head * FM * ECOLS;
  for (int idx = tid; idx < FM * ECOLS; idx += 256)
    atomicAdd(&dst[idx], redLds[idx]);
}

// ---------------------------------------------------------------------------
// Phase B: out = (phi(Q) @ buf1)[:, :64] / (phi(Q) @ buf1)[:, 64]
// grid (64 heads, 16 chunks) x 256 threads; each wave owns 32 rows.
// ---------------------------------------------------------------------------
__global__ __launch_bounds__(256) void perfB(const float* __restrict__ Qg,
                                             const float* __restrict__ buf1,
                                             const float* __restrict__ Og,
                                             float* __restrict__ Out) {
  __shared__ __align__(16) unsigned char pool[8192 + 64*80*2 + 8 * (16*72*2)];
  unsigned short* omegaLds = reinterpret_cast<unsigned short*>(pool);
  unsigned short* b1Lds    = reinterpret_cast<unsigned short*>(pool + 8192);

  const int tid  = threadIdx.x;
  const int lane = tid & 31;
  const int wv   = tid >> 5;
  const int r    = lane & 15;
  const int hf   = lane >> 4;
  const int head = blockIdx.x;

  unsigned short* phiS =
      reinterpret_cast<unsigned short*>(pool + 8192 + 64*80*2) + wv * (16*72);

  for (int idx = tid; idx < DIM * FM; idx += 256)
    omegaLds[idx] = f2bf(Og[idx] * LOG2E);      // exp2 domain
  const float* bsrc = buf1 + (size_t)head * FM * ECOLS;
  for (int idx = tid; idx < FM * 80; idx += 256) {  // [m][80] ushort, pad = 0
    int m = idx / 80, e = idx - m * 80;
    b1Lds[idx] = (e < ECOLS) ? f2bf(bsrc[m * ECOLS + e]) : (unsigned short)0;
  }
  __syncthreads();

  const v8f vzero = {0.f,0.f,0.f,0.f,0.f,0.f,0.f,0.f};
  const float* Qh = Qg + (size_t)head * SEQ * DIM;
  float*       Oh = Out + (size_t)head * SEQ * DIM;

  for (int t = 0; t < 2; ++t) {
    const int s0 = blockIdx.y * 256 + wv * 32 + t * 16;

    if (t == 0)  // prefetch next tile's Q row
      __builtin_prefetch(Qh + (size_t)(s0 + 16 + r) * DIM + 8*hf, 0, 3);

    // ---- phi(Q) for 16 rows, stored row-major [s][m] stride 72 ----
    Frag aQ[2];
    float ss = loadA_row(aQ, Qh + (size_t)(s0 + r) * DIM, hf);
    float rr = fmaf(ss + __shfl_xor(ss, 16, 32), NH_L2E, -3.0f);
    float rh[8];
    #pragma unroll
    for (int i = 0; i < 8; ++i)
      rh[i] = __shfl(rr, i + 8*hf, 32);
    #pragma unroll
    for (int mt = 0; mt < 4; ++mt) {
      Frag b0, b1f;
      loadB(b0,  omegaLds, FM,      8*hf, 16*mt + r);
      loadB(b1f, omegaLds, FM, 32 + 8*hf, 16*mt + r);
      v8f p = vzero;
      p = wmma_bf16(aQ[0], b0,  p);
      p = wmma_bf16(aQ[1], b1f, p);
      unsigned pw[4];
      #pragma unroll
      for (int i = 0; i < 4; ++i)
        pw[i] = pack2(exp2f_dev(p[2*i]   + rh[2*i]),
                      exp2f_dev(p[2*i+1] + rh[2*i+1]));
      // rows are i+8*hf with column 16*mt+r -> 8 scalar bf16 stores
      #pragma unroll
      for (int i = 0; i < 4; ++i) {
        phiS[(2*i     + 8*hf) * 72 + 16*mt + r] = (unsigned short)(pw[i] & 0xFFFFu);
        phiS[(2*i + 1 + 8*hf) * 72 + 16*mt + r] = (unsigned short)(pw[i] >> 16);
      }
    }
    __syncthreads();

    // ---- buf2 tile = phi(Q) (16 x 64) @ buf1 (64 x 80) ----
    Frag aP[2];
    #pragma unroll
    for (int mc = 0; mc < 2; ++mc) {
      const unsigned short* src = phiS + r * 72 + 32*mc + 8*hf;
      aP[mc].q[0] = *reinterpret_cast<const uint4*>(src);
      aP[mc].q[1] = *reinterpret_cast<const uint4*>(src + 16);
    }
    v8f o[5];
    #pragma unroll
    for (int et = 0; et < 5; ++et) {
      o[et] = vzero;
      #pragma unroll
      for (int mc = 0; mc < 2; ++mc) {
        Frag bB;
        loadB(bB, b1Lds, 80, 32*mc + 8*hf, 16*et + r);
        o[et] = wmma_bf16(aP[mc], bB, o[et]);
      }
    }

    // ---- normalize by ones-column (tile 4, col 0 -> lane r==0) & store ----
    #pragma unroll
    for (int i = 0; i < 8; ++i) {
      float den = __shfl(o[4][i], 16*hf, 32);
      float inv = 1.0f / den;
      float* orow = Oh + (size_t)(s0 + i + 8*hf) * DIM + r;
      orow[0]  = o[0][i] * inv;
      orow[16] = o[1][i] * inv;
      orow[32] = o[2][i] * inv;
      orow[48] = o[3][i] * inv;
    }
    __syncthreads();
  }
}

extern "C" void kernel_launch(void* const* d_in, const int* in_sizes, int n_in,
                              void* d_out, int out_size, void* d_ws, size_t ws_size,
                              hipStream_t stream) {
  (void)in_sizes; (void)n_in; (void)out_size; (void)ws_size;
  const float* q  = (const float*)d_in[0];
  const float* k  = (const float*)d_in[1];
  const float* v  = (const float*)d_in[2];
  const float* om = (const float*)d_in[3];
  float* out  = (float*)d_out;
  float* buf1 = (float*)d_ws;  // 64 heads x 64 x 65 f32 = ~1.04 MB scratch

  (void)hipMemsetAsync(buf1, 0, (size_t)NHEAD * FM * ECOLS * sizeof(float), stream);
  perfA<<<dim3(NHEAD, 4),  256, 0, stream>>>(k, v, om, buf1);
  perfB<<<dim3(NHEAD, 16), 256, 0, stream>>>(q, buf1, om, out);
}
